// RHGNLayer_83081847374391
// MI455X (gfx1250) — matrix operand order repaired and checked by reading, not compiled, not run on hardware
//
#include <hip/hip_runtime.h>
#include <hip/hip_bf16.h>

typedef __attribute__((ext_vector_type(16))) _Float16 v16h;
typedef __attribute__((ext_vector_type(8)))  _Float16 v8h;
typedef __attribute__((ext_vector_type(8)))  float    v8f;
typedef __attribute__((ext_vector_type(4)))  unsigned int uint32x4;
typedef __attribute__((ext_vector_type(8)))  unsigned int uint32x8;

#define N_NODE   100000
#define DIM      128
#define N_HEAD   8
#define DKH      16
#define N_TILES  (N_NODE / 16)        // 6250
#define TILES_PER_BLOCK 4             // 4 waves per block, 1 row-tile each

// ---------------------------------------------------------------------------
// WMMA fragment loaders (wave32)
// A layout (16-bit 16x32): lanes 0-15 => M=lane, K = {kb..kb+7, kb+16..kb+23}
//                          lanes 16-31 => M=lane-16, K base += 8
// B layout (16-bit 32x16): lane holds column N=lane&15,
//                          K = {kb..kb+7, kb+16..kb+23}, kb = (lane>>4)*8 (same as A)
// ---------------------------------------------------------------------------
__device__ __forceinline__ v16h load_frag_a_f16(const _Float16* A, int row0, int k0, int lane) {
    int r  = row0 + (lane & 15);
    int kb = k0 + ((lane >> 4) << 3);
    const _Float16* p = A + (size_t)r * DIM + kb;
    v8h lo = *(const v8h*)p;
    v8h hi = *(const v8h*)(p + 16);
    v16h a;
#pragma unroll
    for (int i = 0; i < 8; ++i) { a[i] = lo[i]; a[8 + i] = hi[i]; }
    return a;
}

__device__ __forceinline__ v16h load_frag_b_lds(const _Float16* shBT, int col0, int k0, int lane) {
    int c  = col0 + (lane & 15);
    int kb = k0 + ((lane >> 4) << 3);
    const _Float16* p = shBT + (size_t)c * DIM + kb;
    v8h lo = *(const v8h*)p;
    v8h hi = *(const v8h*)(p + 16);
    v16h b;
#pragma unroll
    for (int i = 0; i < 8; ++i) { b[i] = lo[i]; b[8 + i] = hi[i]; }
    return b;
}

// ---------------------------------------------------------------------------
// TDM: stage a 128x128 f16 matrix (row-major, contiguous) into LDS.
// D# per CDNA5 ISA:
//  group0: [1:0]=count=1, [63:32]=lds_addr, [120:64]=global_addr, [127:126]=type=2
//  group1: [17:16]=data_size=1 (2B), [79:48]=tensor_dim0=128, [111:80]=tensor_dim1=128,
//          [127:112]=tile_dim0=128, [143:128]=tile_dim1=128, [207:160]=dim0_stride=128
// ---------------------------------------------------------------------------
__device__ __forceinline__ void tdm_load_128x128_f16(const _Float16* gsrc, _Float16* lds_dst) {
    unsigned long long ga = (unsigned long long)(uintptr_t)gsrc;
    unsigned lds_off = (unsigned)(uintptr_t)lds_dst;   // low 32 bits of LDS aperture = offset
    uint32x4 g0;
    g0[0] = 1u;                                        // count=1 valid descriptor
    g0[1] = lds_off;                                   // lds_addr
    g0[2] = (unsigned)(ga & 0xFFFFFFFFu);              // global_addr[31:0]
    g0[3] = (unsigned)((ga >> 32) & 0x1FFFFFFu) | (2u << 30);  // global_addr[56:32] | type=2
    uint32x8 g1;
    g1[0] = 1u << 16;                                  // data_size = 1 (2 bytes)
    g1[1] = 128u << 16;                                // tensor_dim0[15:0] in bits[63:48]
    g1[2] = 128u << 16;                                // tensor_dim0 msbs=0, tensor_dim1=128 (bits[111:80])
    g1[3] = 128u << 16;                                // tensor_dim1 msbs=0, tile_dim0=128 (bits[127:112])
    g1[4] = 128u;                                      // tile_dim1=128, tile_dim2=0
    g1[5] = 128u;                                      // tensor_dim0_stride[31:0] = 128
    g1[6] = 0u;                                        // dim0_stride msbs, dim1_stride lo
    g1[7] = 0u;                                        // dim1_stride hi
    asm volatile("tensor_load_to_lds %0, %1" :: "s"(g0), "s"(g1) : "memory");
}

// ---------------------------------------------------------------------------
// f32 -> f16 conversion for h
// ---------------------------------------------------------------------------
__global__ void conv_f32_f16_kernel(const float* __restrict__ in, _Float16* __restrict__ out, int n) {
    int i = blockIdx.x * blockDim.x + threadIdx.x;
    if (i < n) out[i] = (_Float16)in[i];
}

// ---------------------------------------------------------------------------
// Fold rel_att/rel_msg into W and produce transposed f16 weights.
// blocks [0,384): r,j  -> WkT_eff / WvT_eff / biases
// blocks [384,640): t,j -> WqT / WaT transposes
// ---------------------------------------------------------------------------
__global__ void fold_weights_kernel(const float* __restrict__ Wk, const float* __restrict__ bk,
                                    const float* __restrict__ Wq,
                                    const float* __restrict__ Wv, const float* __restrict__ bv,
                                    const float* __restrict__ Wa,
                                    const float* __restrict__ rel_att, const float* __restrict__ rel_msg,
                                    _Float16* __restrict__ WkT_eff, float* __restrict__ bk_eff,
                                    _Float16* __restrict__ WvT_eff, float* __restrict__ bv_eff,
                                    _Float16* __restrict__ WqT, _Float16* __restrict__ WaT) {
    int bx = blockIdx.x;
    int k  = threadIdx.x;   // 0..127
    if (bx < 384) {
        int r = bx >> 7, j = bx & 127;
        int h = j >> 4, dj = j & 15;
        int sid = (r == 2) ? 1 : 0;
        const float* Wkm = Wk + (size_t)sid * DIM * DIM;
        const float* Wvm = Wv + (size_t)sid * DIM * DIM;
        const float* Ra  = rel_att + ((size_t)(r * N_HEAD + h)) * DKH * DKH;
        const float* Rm  = rel_msg + ((size_t)(r * N_HEAD + h)) * DKH * DKH;
        float ak = 0.f, av = 0.f;
#pragma unroll
        for (int d = 0; d < DKH; ++d) {
            ak += Wkm[k * DIM + h * DKH + d] * Ra[d * DKH + dj];
            av += Wvm[k * DIM + h * DKH + d] * Rm[d * DKH + dj];
        }
        WkT_eff[((size_t)r * DIM + j) * DIM + k] = (_Float16)ak;
        WvT_eff[((size_t)r * DIM + j) * DIM + k] = (_Float16)av;
        if (k == 0) {
            const float* bkm = bk + sid * DIM;
            const float* bvm = bv + sid * DIM;
            float sk = 0.f, sv = 0.f;
#pragma unroll
            for (int d = 0; d < DKH; ++d) {
                sk += bkm[h * DKH + d] * Ra[d * DKH + dj];
                sv += bvm[h * DKH + d] * Rm[d * DKH + dj];
            }
            bk_eff[r * DIM + j] = sk;
            bv_eff[r * DIM + j] = sv;
        }
    } else {
        int bx2 = bx - 384;
        int t = bx2 >> 7, j = bx2 & 127;
        WqT[((size_t)t * DIM + j) * DIM + k] = (_Float16)Wq[(size_t)t * DIM * DIM + k * DIM + j];
        WaT[((size_t)t * DIM + j) * DIM + k] = (_Float16)Wa[(size_t)t * DIM * DIM + k * DIM + j];
    }
}

// ---------------------------------------------------------------------------
// Node transform GEMM: out[n][j] = f16( sum_k A[n][k]*BT[j][k] + bias[j] )
// Block = 128 threads (4 waves). Wave 0 stages BT (32KB) into LDS via TDM.
// Each wave computes one 16-row x 128-col strip: 32 WMMAs, A regs loaded once.
// ---------------------------------------------------------------------------
__global__ void gemm_node_kernel(const _Float16* __restrict__ A, const _Float16* __restrict__ BT,
                                 const float* __restrict__ bias, _Float16* __restrict__ out) {
    __shared__ _Float16 shBT[DIM * DIM];
    int lane = threadIdx.x & 31;
    int wave = threadIdx.x >> 5;
    int row_tile = blockIdx.x * TILES_PER_BLOCK + wave;
    bool valid = row_tile < N_TILES;
    int row0 = row_tile << 4;

    v16h a0, a1, a2, a3;
    if (valid) {
        a0 = load_frag_a_f16(A, row0, 0,  lane);
        a1 = load_frag_a_f16(A, row0, 32, lane);
        a2 = load_frag_a_f16(A, row0, 64, lane);
        a3 = load_frag_a_f16(A, row0, 96, lane);
    }
    if (wave == 0) {
        tdm_load_128x128_f16(BT, shBT);
        __builtin_amdgcn_s_wait_tensorcnt(0);
    }
    __syncthreads();
    if (!valid) return;

#pragma unroll
    for (int t = 0; t < 8; ++t) {
        int col0 = t << 4;
        v8f acc = {};
        acc = __builtin_amdgcn_wmma_f32_16x16x32_f16(false, a0, false, load_frag_b_lds(shBT, col0, 0,  lane), (short)0, acc, false, false);
        acc = __builtin_amdgcn_wmma_f32_16x16x32_f16(false, a1, false, load_frag_b_lds(shBT, col0, 32, lane), (short)0, acc, false, false);
        acc = __builtin_amdgcn_wmma_f32_16x16x32_f16(false, a2, false, load_frag_b_lds(shBT, col0, 64, lane), (short)0, acc, false, false);
        acc = __builtin_amdgcn_wmma_f32_16x16x32_f16(false, a3, false, load_frag_b_lds(shBT, col0, 96, lane), (short)0, acc, false, false);
        int n = col0 + (lane & 15);
        float bcol = bias[n];
#pragma unroll
        for (int g = 0; g < 8; ++g) {
            int m = row0 + g + ((lane >> 4) << 3);
            out[(size_t)m * DIM + n] = (_Float16)(acc[g] + bcol);
        }
    }
}

// ---------------------------------------------------------------------------
// Edge attention: e = exp(q[dst].k[src] * pri / sqrt(16)); accumulate denom.
// Softmax max-subtraction omitted (mathematically identical, scores O(1)).
// ---------------------------------------------------------------------------
__global__ void edge_att_kernel(const _Float16* __restrict__ K, const _Float16* __restrict__ Q,
                                const int* __restrict__ src, const int* __restrict__ dst,
                                const float* __restrict__ pri, float* __restrict__ att,
                                float* __restrict__ denom, int nE) {
    int e = blockIdx.x * blockDim.x + threadIdx.x;
    if (e >= nE) return;
    int s = src[e], d = dst[e];
    const v16h* kp = (const v16h*)(K + (size_t)s * DIM);
    const v16h* qp = (const v16h*)(Q + (size_t)d * DIM);
#pragma unroll
    for (int h = 0; h < N_HEAD; ++h) {
        v16h kv = kp[h];
        v16h qv = qp[h];
        float acc = 0.f;
#pragma unroll
        for (int i = 0; i < DKH; ++i)
            acc += (float)kv[i] * (float)qv[i];
        float ex = __expf(acc * pri[h] * 0.25f);
        att[(size_t)e * N_HEAD + h] = ex;
        atomicAdd(&denom[(size_t)d * N_HEAD + h], ex);
    }
}

// ---------------------------------------------------------------------------
// Edge aggregation: agg[dst] += v[src] * (att/denom[dst]) per head
// ---------------------------------------------------------------------------
__global__ void edge_agg_kernel(const _Float16* __restrict__ V,
                                const int* __restrict__ src, const int* __restrict__ dst,
                                const float* __restrict__ att, const float* __restrict__ denom,
                                float* __restrict__ agg, int nE) {
    int e = blockIdx.x * blockDim.x + threadIdx.x;
    if (e >= nE) return;
    int s = src[e], d = dst[e];
    const v16h* vp = (const v16h*)(V + (size_t)s * DIM);
#pragma unroll
    for (int h = 0; h < N_HEAD; ++h) {
        float w = att[(size_t)e * N_HEAD + h] / denom[(size_t)d * N_HEAD + h];
        v16h vv = vp[h];
#pragma unroll
        for (int i = 0; i < DKH; ++i)
            atomicAdd(&agg[(size_t)d * DIM + h * DKH + i], (float)vv[i] * w);
    }
}

// ---------------------------------------------------------------------------
// Final GEMM + skip mix:
// out = (agg*scale @ Wa + ba) * sigmoid(skip) + h * (1 - sigmoid(skip))
// Same structure as gemm_node: TDM-staged WaT in LDS, 32 WMMAs per wave.
// ---------------------------------------------------------------------------
__global__ void gemm_final_kernel(const float* __restrict__ Aagg, float scale,
                                  const _Float16* __restrict__ BT, const float* __restrict__ bias,
                                  const float* __restrict__ skip, const float* __restrict__ hres,
                                  float* __restrict__ out) {
    __shared__ _Float16 shBT[DIM * DIM];
    int lane = threadIdx.x & 31;
    int wave = threadIdx.x >> 5;
    int row_tile = blockIdx.x * TILES_PER_BLOCK + wave;
    bool valid = row_tile < N_TILES;
    int row0 = row_tile << 4;

    v16h a[4];
    if (valid) {
#pragma unroll
        for (int c = 0; c < 4; ++c) {
            int r  = row0 + (lane & 15);
            int kb = 32 * c + ((lane >> 4) << 3);
            const float* p = Aagg + (size_t)r * DIM + kb;
            v8f lo = *(const v8f*)p;
            v8f hi = *(const v8f*)(p + 16);
#pragma unroll
            for (int i = 0; i < 8; ++i) {
                a[c][i]     = (_Float16)(lo[i] * scale);
                a[c][8 + i] = (_Float16)(hi[i] * scale);
            }
        }
    }
    if (wave == 0) {
        tdm_load_128x128_f16(BT, shBT);
        __builtin_amdgcn_s_wait_tensorcnt(0);
    }
    __syncthreads();
    if (!valid) return;

    float alpha = 1.0f / (1.0f + __expf(-skip[0]));
#pragma unroll
    for (int t = 0; t < 8; ++t) {
        int col0 = t << 4;
        v8f acc = {};
#pragma unroll
        for (int c = 0; c < 4; ++c)
            acc = __builtin_amdgcn_wmma_f32_16x16x32_f16(false, a[c], false,
                      load_frag_b_lds(shBT, col0, 32 * c, lane), (short)0, acc, false, false);
        int n = col0 + (lane & 15);
        float bcol = bias[n];
#pragma unroll
        for (int g = 0; g < 8; ++g) {
            int m = row0 + g + ((lane >> 4) << 3);
            float trans = acc[g] + bcol;
            out[(size_t)m * DIM + n] = trans * alpha + hres[(size_t)m * DIM + n] * (1.0f - alpha);
        }
    }
}

// ---------------------------------------------------------------------------
extern "C" void kernel_launch(void* const* d_in, const int* in_sizes, int n_in,
                              void* d_out, int out_size, void* d_ws, size_t ws_size,
                              hipStream_t stream) {
    const float* h_item   = (const float*)d_in[0];
    const float* h_user   = (const float*)d_in[1];
    const float* Wk       = (const float*)d_in[2];
    const float* bk       = (const float*)d_in[3];
    const float* Wq       = (const float*)d_in[4];
    const float* bq       = (const float*)d_in[5];
    const float* Wv       = (const float*)d_in[6];
    const float* bv       = (const float*)d_in[7];
    const float* Wa       = (const float*)d_in[8];
    const float* ba       = (const float*)d_in[9];
    const float* rel_pri  = (const float*)d_in[10];
    const float* rel_att  = (const float*)d_in[11];
    const float* rel_msg  = (const float*)d_in[12];
    const float* skip     = (const float*)d_in[13];
    const int* click_src  = (const int*)d_in[14];
    const int* click_dst  = (const int*)d_in[15];
    const int* purch_src  = (const int*)d_in[16];
    const int* purch_dst  = (const int*)d_in[17];
    const int* rev_src    = (const int*)d_in[18];
    const int* rev_dst    = (const int*)d_in[19];

    const int nClick = in_sizes[14], nPurch = in_sizes[16], nRev = in_sizes[18];

    // ---- workspace carving (all chunks 256B-aligned by construction) ----
    const size_t NODE_F16 = (size_t)N_NODE * DIM * sizeof(_Float16);   // 25.6 MB
    const size_t NODE_F32 = (size_t)N_NODE * DIM * sizeof(float);      // 51.2 MB
    char* w = (char*)d_ws;
    auto take = [&](size_t sz) { char* p = w; w += sz; return p; };

    _Float16* h_item16 = (_Float16*)take(NODE_F16);
    _Float16* h_user16 = (_Float16*)take(NODE_F16);
    _Float16* k0 = (_Float16*)take(NODE_F16);
    _Float16* k1 = (_Float16*)take(NODE_F16);
    _Float16* k2 = (_Float16*)take(NODE_F16);
    _Float16* v0 = (_Float16*)take(NODE_F16);
    _Float16* v1 = (_Float16*)take(NODE_F16);
    _Float16* v2 = (_Float16*)take(NODE_F16);
    _Float16* q_item = (_Float16*)take(NODE_F16);
    _Float16* q_user = (_Float16*)take(NODE_F16);
    _Float16* WkT_eff = (_Float16*)take((size_t)3 * DIM * DIM * sizeof(_Float16));
    _Float16* WvT_eff = (_Float16*)take((size_t)3 * DIM * DIM * sizeof(_Float16));
    _Float16* WqT     = (_Float16*)take((size_t)2 * DIM * DIM * sizeof(_Float16));
    _Float16* WaT     = (_Float16*)take((size_t)2 * DIM * DIM * sizeof(_Float16));
    float* bk_eff = (float*)take((size_t)3 * DIM * sizeof(float));
    float* bv_eff = (float*)take((size_t)3 * DIM * sizeof(float) + 2560); // pad
    float* att_all    = (float*)take((size_t)(nClick + nPurch + nRev) * N_HEAD * sizeof(float));
    float* denom_user = (float*)take((size_t)N_NODE * N_HEAD * sizeof(float));
    float* denom_item = (float*)take((size_t)N_NODE * N_HEAD * sizeof(float));
    float* agg_item   = (float*)take(NODE_F32);
    float* agg_user   = (float*)take(NODE_F32);

    float* att_click = att_all;
    float* att_purch = att_all + (size_t)nClick * N_HEAD;
    float* att_rev   = att_all + (size_t)(nClick + nPurch) * N_HEAD;

    // ---- zero accumulators (graph-capture-safe) ----
    hipMemsetAsync(denom_user, 0, (size_t)N_NODE * N_HEAD * sizeof(float), stream);
    hipMemsetAsync(denom_item, 0, (size_t)N_NODE * N_HEAD * sizeof(float), stream);
    hipMemsetAsync(agg_item, 0, NODE_F32, stream);
    hipMemsetAsync(agg_user, 0, NODE_F32, stream);

    // ---- phase 0: conversions + weight folding ----
    {
        int n = N_NODE * DIM;
        int blocks = (n + 255) / 256;
        conv_f32_f16_kernel<<<blocks, 256, 0, stream>>>(h_item, h_item16, n);
        conv_f32_f16_kernel<<<blocks, 256, 0, stream>>>(h_user, h_user16, n);
    }
    fold_weights_kernel<<<640, 128, 0, stream>>>(Wk, bk, Wq, Wv, bv, Wa, rel_att, rel_msg,
                                                 WkT_eff, bk_eff, WvT_eff, bv_eff, WqT, WaT);

    // ---- phase 1: node transform GEMMs (WMMA + TDM-staged weights) ----
    int gblocks = (N_TILES + TILES_PER_BLOCK - 1) / TILES_PER_BLOCK;   // 1563
    gemm_node_kernel<<<gblocks, 128, 0, stream>>>(h_item16, WkT_eff + 0 * DIM * DIM, bk_eff + 0 * DIM, k0);
    gemm_node_kernel<<<gblocks, 128, 0, stream>>>(h_item16, WkT_eff + 1 * DIM * DIM, bk_eff + 1 * DIM, k1);
    gemm_node_kernel<<<gblocks, 128, 0, stream>>>(h_user16, WkT_eff + 2 * DIM * DIM, bk_eff + 2 * DIM, k2);
    gemm_node_kernel<<<gblocks, 128, 0, stream>>>(h_item16, WvT_eff + 0 * DIM * DIM, bv_eff + 0 * DIM, v0);
    gemm_node_kernel<<<gblocks, 128, 0, stream>>>(h_item16, WvT_eff + 1 * DIM * DIM, bv_eff + 1 * DIM, v1);
    gemm_node_kernel<<<gblocks, 128, 0, stream>>>(h_user16, WvT_eff + 2 * DIM * DIM, bv_eff + 2 * DIM, v2);
    gemm_node_kernel<<<gblocks, 128, 0, stream>>>(h_item16, WqT + 0 * DIM * DIM, bq + 0 * DIM, q_item);
    gemm_node_kernel<<<gblocks, 128, 0, stream>>>(h_user16, WqT + 1 * DIM * DIM, bq + 1 * DIM, q_user);

    // ---- phase 2: edge attention (exp + denom) ----
    edge_att_kernel<<<(nClick + 255) / 256, 256, 0, stream>>>(k0, q_user, click_src, click_dst,
                                                              rel_pri + 0 * N_HEAD, att_click, denom_user, nClick);
    edge_att_kernel<<<(nPurch + 255) / 256, 256, 0, stream>>>(k1, q_user, purch_src, purch_dst,
                                                              rel_pri + 1 * N_HEAD, att_purch, denom_user, nPurch);
    edge_att_kernel<<<(nRev + 255) / 256, 256, 0, stream>>>(k2, q_item, rev_src, rev_dst,
                                                            rel_pri + 2 * N_HEAD, att_rev, denom_item, nRev);

    // ---- phase 3: weighted aggregation ----
    edge_agg_kernel<<<(nClick + 255) / 256, 256, 0, stream>>>(v0, click_src, click_dst,
                                                              att_click, denom_user, agg_user, nClick);
    edge_agg_kernel<<<(nPurch + 255) / 256, 256, 0, stream>>>(v1, purch_src, purch_dst,
                                                              att_purch, denom_user, agg_user, nPurch);
    edge_agg_kernel<<<(nRev + 255) / 256, 256, 0, stream>>>(v2, rev_src, rev_dst,
                                                            att_rev, denom_item, agg_item, nRev);

    // ---- phase 4: output GEMM + skip mix (WMMA + TDM-staged weights) ----
    float* out = (float*)d_out;
    gemm_final_kernel<<<gblocks, 128, 0, stream>>>(agg_item, 1.0f, WaT + 0 * DIM * DIM, ba + 0 * DIM,
                                                   skip + 0, h_item, out);
    gemm_final_kernel<<<gblocks, 128, 0, stream>>>(agg_user, 0.5f, WaT + 1 * DIM * DIM, ba + 1 * DIM,
                                                   skip + 1, h_user, out + (size_t)N_NODE * DIM);
}
